// Seq2seq_73117523247567
// MI455X (gfx1250) — matrix operand order repaired and checked by reading, not compiled, not run on hardware
//
#include <hip/hip_runtime.h>
#include <hip/hip_bf16.h>

typedef _Float16 f16;
typedef __attribute__((ext_vector_type(16))) _Float16 v16h;
typedef __attribute__((ext_vector_type(8)))  _Float16 v8h;
typedef __attribute__((ext_vector_type(8)))  float    v8f;

// ---------------- workspace layout (halves) ----------------
// 0     : embed_in  f16 [96][32]            (3072)
// 3072  : embed_dec f16 [96][32]            (3072)
// 6144  : encoder W fragments (16 frags * 512)   [kc(2) x nt(8)]  K=[Wk;Wr]
// 14336 : decoder W fragments (16 frags * 512)
// 22528 : W1 fragments (4 frags * 512)      [nt(4)], K=32
// 24576 : W2 fragments (12 frags * 512)     [kc(2) x nt(6)], K=64
// total 30720 halves = 61440 bytes of d_ws
#define WS_EMB_IN   0
#define WS_EMB_DEC  3072
#define WS_ENCW     6144
#define WS_DECW     14336
#define WS_W1       22528
#define WS_W2       24576
#define WS_TOTAL    30720

#define LOG2E 1.4426950408889634f

// Branchless hardware transcendentals: single v_exp_f32 / v_rcp_f32 TRANS ops.
__device__ __forceinline__ float fast_exp2(float x) { return __builtin_amdgcn_exp2f(x); }
__device__ __forceinline__ float fast_rcp(float x)  { return __builtin_amdgcn_rcpf(x); }
__device__ __forceinline__ float fast_sigmoid(float x) {
  return fast_rcp(1.f + fast_exp2(-LOG2E * x));
}
__device__ __forceinline__ float fast_tanh(float x) {
  // tanh(x) = 1 - 2/(exp(2x)+1); saturates to +-1 without branches
  float s = fast_exp2(2.f * LOG2E * x);
  return 1.f - 2.f * fast_rcp(s + 1.f);
}

__device__ __forceinline__ v8f wmma_f16(v16h a, v16h b, v8f c) {
  // D = A(16x32 f16) * B(32x16 f16) + C(16x16 f32)
  return __builtin_amdgcn_wmma_f32_16x16x32_f16(
      /*neg_a=*/false, a, /*neg_b=*/false, b,
      /*c_mod=*/(short)0, c, /*reuse_a=*/false, /*reuse_b=*/false);
}

// A fragment: lane holds row m=lane&15; halves j0..15 = K k0..k0+15,
// k0 = hi?16:0.  rowptr points at 32 contiguous halves.
__device__ __forceinline__ v16h load_rowA(const f16* rowptr, int hi) {
  const int k0 = hi ? 16 : 0;
  v8h a = *(const v8h*)(rowptr + k0);
  v8h b = *(const v8h*)(rowptr + k0 + 8);
  v16h r;
#pragma unroll
  for (int j = 0; j < 8; ++j) { r[j] = a[j]; r[j + 8] = b[j]; }
  return r;
}

// B fragment: packed lane-major, 16 halves per lane, contiguous.
__device__ __forceinline__ v16h load_bfrag(const f16* fragbase, int ln) {
  v8h a = *(const v8h*)(fragbase + ln * 16);
  v8h b = *(const v8h*)(fragbase + ln * 16 + 8);
  v16h r;
#pragma unroll
  for (int j = 0; j < 8; ++j) { r[j] = a[j]; r[j + 8] = b[j]; }
  return r;
}

// ---------------- prep: f32 -> f16 fragment packing ----------------
__global__ void prep_kernel(const float* __restrict__ eIn, const float* __restrict__ eDec,
                            const float* __restrict__ encK, const float* __restrict__ encR,
                            const float* __restrict__ decK, const float* __restrict__ decR,
                            const float* __restrict__ W1, const float* __restrict__ W2,
                            f16* __restrict__ ws) {
  int p = blockIdx.x * blockDim.x + threadIdx.x;
  if (p >= WS_TOTAL) return;
  float v;
  if (p < 3072) {
    v = eIn[p];                                  // [96][32] row-major
  } else if (p < 6144) {
    v = eDec[p - 3072];
  } else if (p < WS_W1) {
    int q = p - WS_ENCW;                         // 0..16383 : enc then dec LSTM weights
    const float *K = encK, *R = encR;
    if (q >= 8192) { K = decK; R = decR; q -= 8192; }
    int frag = q >> 9, ln = (q >> 4) & 31, j = q & 15;
    int kc = frag >> 3, nt = frag & 7;
    int n  = nt * 16 + (ln & 15);
    int kl = (ln < 16) ? j : 16 + j;             // B layout: lanes 0-15 K0..15, 16-31 K16..31
    v = (kc == 0) ? K[kl * 128 + n] : R[kl * 128 + n];
  } else if (p < WS_W2) {
    int q = p - WS_W1;
    int frag = q >> 9, ln = (q >> 4) & 31, j = q & 15;
    int n  = frag * 16 + (ln & 15);
    int kl = (ln < 16) ? j : 16 + j;
    v = W1[kl * 64 + n];                         // [32][64]
  } else {
    int q = p - WS_W2;
    int frag = q >> 9, ln = (q >> 4) & 31, j = q & 15;
    int kc = frag / 6, nt = frag % 6;
    int n = nt * 16 + (ln & 15);
    int k = kc * 32 + ((ln < 16) ? j : 16 + j);
    v = W2[k * 96 + n];                          // [64][96]
  }
  ws[p] = (f16)v;
}

// ---------------- fused encoder+decoder ----------------
// 64 WGs x 128 threads (4 waves). Wave owns 16 batch rows for all 128 steps.
#define HSTR 40   // padded row stride (halves) for h buffer (bank spread, 16B aligned)
#define YSTR 72   // padded row stride for y buffer

__device__ __forceinline__ void lstm_gemm(const v16h& x, const v16h& h,
                                          const f16* Wb, const float* bias,
                                          int mrow, int ln, v8f* z) {
#pragma unroll
  for (int nt = 0; nt < 8; ++nt) {
    float bv = bias[nt * 16 + mrow];
    v8f acc;
#pragma unroll
    for (int r = 0; r < 8; ++r) acc[r] = bv;
    acc = wmma_f16(x, load_bfrag(Wb + nt * 512, ln), acc);         // x @ Wk
    acc = wmma_f16(h, load_bfrag(Wb + (8 + nt) * 512, ln), acc);   // h @ Wr
    z[nt] = acc;
  }
}

__device__ __forceinline__ void lstm_update(const v8f* z, v8f& c0, v8f& c1,
                                            f16* hbw, int mrow, int hi) {
#pragma unroll
  for (int j = 0; j < 2; ++j) {                 // two 16-col tiles of H=32
    v8f& c = j ? c1 : c0;
#pragma unroll
    for (int r = 0; r < 8; ++r) {
      float iv = fast_sigmoid(z[0 + j][r]);     // gate order i,f,g,o over 4H
      float fv = fast_sigmoid(z[2 + j][r]);
      float gv = fast_tanh(z[4 + j][r]);
      float ov = fast_sigmoid(z[6 + j][r]);
      float cv = fv * c[r] + iv * gv;
      c[r] = cv;
      hbw[(r + 8 * hi) * HSTR + j * 16 + mrow] = (f16)(ov * fast_tanh(cv));
    }
  }
}

__global__ __launch_bounds__(128) void seq2seq_kernel(
    const int* __restrict__ inputs, const int* __restrict__ targets,
    const float* __restrict__ enc_b, const float* __restrict__ dec_b,
    const float* __restrict__ b1, const float* __restrict__ b2,
    const f16* __restrict__ ws, float* __restrict__ out) {
  __shared__ __align__(16) f16 ebuf[3072];        // embed table (phase-swapped)
  __shared__ __align__(16) f16 wbuf[8192];        // LSTM W fragments (phase-swapped)
  __shared__ __align__(16) f16 hbuf[4][16 * HSTR];
  __shared__ __align__(16) f16 ybuf[4][16 * YSTR];
  __shared__ float bb[416];                       // enc_b(128) dec_b(128) b1(64) b2(96)

  const int tid  = threadIdx.x;
  const int wv   = tid >> 5;
  const int ln   = tid & 31;
  const int mrow = ln & 15;
  const int hi   = ln >> 4;
  const int row0 = blockIdx.x * 64 + wv * 16;

  if (tid < 128) { bb[tid] = enc_b[tid]; bb[128 + tid] = dec_b[tid]; }
  if (tid < 64)  bb[256 + tid] = b1[tid];
  if (tid < 96)  bb[320 + tid] = b2[tid];
  {
    const uint4* s1 = (const uint4*)(ws + WS_EMB_IN);
    const uint4* s2 = (const uint4*)(ws + WS_ENCW);
    uint4* d1 = (uint4*)ebuf; uint4* d2 = (uint4*)wbuf;
    for (int i = tid; i < 384;  i += 128) d1[i] = s1[i];
    for (int i = tid; i < 1024; i += 128) d2[i] = s2[i];
  }
  __syncthreads();

  f16* hbw = hbuf[wv];
  f16* ybw = ybuf[wv];

  v8f c0, c1;
#pragma unroll
  for (int r = 0; r < 8; ++r) { c0[r] = 0.f; c1[r] = 0.f; }
  v16h hfrag;
#pragma unroll
  for (int j = 0; j < 16; ++j) hfrag[j] = (f16)0.f;

  // ---------------- encoder ----------------
  for (int t = 0; t < 64; ++t) {
    int tok = inputs[(row0 + mrow) * 64 + t];
    v16h xfrag = load_rowA(ebuf + tok * 32, hi);
    v8f z[8];
    lstm_gemm(xfrag, hfrag, wbuf, bb, mrow, ln, z);
    lstm_update(z, c0, c1, hbw, mrow, hi);
    hfrag = load_rowA(hbw + mrow * HSTR, hi);     // C-layout -> A-layout via LDS
  }

  // swap LDS tables: decoder embed + decoder LSTM weights
  __syncthreads();
  {
    const uint4* s1 = (const uint4*)(ws + WS_EMB_DEC);
    const uint4* s2 = (const uint4*)(ws + WS_DECW);
    uint4* d1 = (uint4*)ebuf; uint4* d2 = (uint4*)wbuf;
    for (int i = tid; i < 384;  i += 128) d1[i] = s1[i];
    for (int i = tid; i < 1024; i += 128) d2[i] = s2[i];
  }
  __syncthreads();

  const f16* W1f = ws + WS_W1;   // dense weights stay in global (L2-resident, b128 loads)
  const f16* W2f = ws + WS_W2;

  // ---------------- decoder ----------------
  for (int t = 0; t < 64; ++t) {
    int tok = targets[(row0 + mrow) * 64 + t];
    v16h xfrag = load_rowA(ebuf + tok * 32, hi);
    v8f z[8];
    lstm_gemm(xfrag, hfrag, wbuf, bb + 128, mrow, ln, z);
    lstm_update(z, c0, c1, hbw, mrow, hi);
    hfrag = load_rowA(hbw + mrow * HSTR, hi);

    // y1 = relu(h @ W1 + b1)  [16x64]
    v8f y[4];
#pragma unroll
    for (int nt = 0; nt < 4; ++nt) {
      float bv = bb[256 + nt * 16 + mrow];
      v8f acc;
#pragma unroll
      for (int r = 0; r < 8; ++r) acc[r] = bv;
      y[nt] = wmma_f16(hfrag, load_bfrag(W1f + nt * 512, ln), acc);
    }
#pragma unroll
    for (int nt = 0; nt < 4; ++nt)
#pragma unroll
      for (int r = 0; r < 8; ++r) {
        float v = y[nt][r];
        ybw[(r + 8 * hi) * YSTR + nt * 16 + mrow] = (f16)(v > 0.f ? v : 0.f);
      }
    v16h ya0 = load_rowA(ybw + mrow * YSTR, hi);        // K chunk 0..31
    v16h ya1 = load_rowA(ybw + mrow * YSTR + 32, hi);   // K chunk 32..63

    // logits = y1 @ W2 + b2  [16x96]
    v8f lg[6];
#pragma unroll
    for (int nt = 0; nt < 6; ++nt) {
      float bv = bb[320 + nt * 16 + mrow];
      v8f acc;
#pragma unroll
      for (int r = 0; r < 8; ++r) acc[r] = bv;
      acc = wmma_f16(ya0, load_bfrag(W2f + nt * 512, ln), acc);
      acc = wmma_f16(ya1, load_bfrag(W2f + (6 + nt) * 512, ln), acc);
      lg[nt] = acc;
    }

    // row softmax: row m's 96 values live in VGPR (m&7) across 16 lanes x 6 tiles.
    float rmax[8], rrcp[8];
#pragma unroll
    for (int r = 0; r < 8; ++r) {
      float m = lg[0][r];
#pragma unroll
      for (int nt = 1; nt < 6; ++nt) m = fmaxf(m, lg[nt][r]);
#pragma unroll
      for (int s = 1; s < 16; s <<= 1) m = fmaxf(m, __shfl_xor(m, s, 32));
      rmax[r] = m;
    }
    // exp once per element (reused for sum and output)
#pragma unroll
    for (int nt = 0; nt < 6; ++nt)
#pragma unroll
      for (int r = 0; r < 8; ++r)
        lg[nt][r] = fast_exp2((lg[nt][r] - rmax[r]) * LOG2E);
#pragma unroll
    for (int r = 0; r < 8; ++r) {
      float ssum = 0.f;
#pragma unroll
      for (int nt = 0; nt < 6; ++nt) ssum += lg[nt][r];
#pragma unroll
      for (int s = 1; s < 16; s <<= 1) ssum += __shfl_xor(ssum, s, 32);
      rrcp[r] = fast_rcp(ssum);
    }

    float* ob = out + ((size_t)t * 4096 + row0) * 96;
#pragma unroll
    for (int nt = 0; nt < 6; ++nt)
#pragma unroll
      for (int r = 0; r < 8; ++r)
        ob[(r + 8 * hi) * 96 + nt * 16 + mrow] = lg[nt][r] * rrcp[r];
  }
}

extern "C" void kernel_launch(void* const* d_in, const int* in_sizes, int n_in,
                              void* d_out, int out_size, void* d_ws, size_t ws_size,
                              hipStream_t stream) {
  const int*   inputs    = (const int*)  d_in[0];
  const int*   targets   = (const int*)  d_in[1];
  const float* embed_in  = (const float*)d_in[2];
  const float* embed_dec = (const float*)d_in[3];
  const float* enc_k     = (const float*)d_in[4];
  const float* enc_rk    = (const float*)d_in[5];
  const float* enc_b     = (const float*)d_in[6];
  const float* dec_k     = (const float*)d_in[7];
  const float* dec_rk    = (const float*)d_in[8];
  const float* dec_b     = (const float*)d_in[9];
  const float* W1        = (const float*)d_in[10];
  const float* b1        = (const float*)d_in[11];
  const float* W2        = (const float*)d_in[12];
  const float* b2        = (const float*)d_in[13];
  f16* wsh = (f16*)d_ws;   // needs 61,440 bytes

  prep_kernel<<<(WS_TOTAL + 127) / 128, 128, 0, stream>>>(
      embed_in, embed_dec, enc_k, enc_rk, dec_k, dec_rk, W1, W2, wsh);
  seq2seq_kernel<<<64, 128, 0, stream>>>(
      inputs, targets, enc_b, dec_b, b1, b2, wsh, (float*)d_out);
}
